// TransformerSelfAttentionRing_24051816857738
// MI455X (gfx1250) — matrix operand order repaired and checked by reading, not compiled, not run on hardware
//
#include <hip/hip_runtime.h>
#include <hip/hip_bf16.h>
#include <stdint.h>
#include <stddef.h>

// Problem dimensions (fixed by the reference).
#define S_DIM   2048
#define B_DIM   2
#define H_DIM   2048
#define NH_DIM  16
#define HD_DIM  128
#define QKV_N   (3 * NH_DIM * HD_DIM)   // 6144
#define PROJ_N  (NH_DIM * HD_DIM)       // 2048
#define M_ROWS  (S_DIM * B_DIM)         // 4096
#define QKV_ROW (NH_DIM * 3 * HD_DIM)   // 6144 elements per (s,b) row
#define NORM_INV 0.02209708691207961f   // 1/sqrt(2048)

typedef __attribute__((ext_vector_type(16))) __bf16       v16bf;
typedef __attribute__((ext_vector_type(8)))  __bf16       v8bf;
typedef __attribute__((ext_vector_type(8)))  float        v8f;
typedef __attribute__((ext_vector_type(4)))  unsigned int u32x4;

// ---------------------------------------------------------------------------
// helpers
// ---------------------------------------------------------------------------
__device__ __forceinline__ v16bf bfcat(v8bf a, v8bf b) {
  return __builtin_shufflevector(a, b, 0,1,2,3,4,5,6,7,8,9,10,11,12,13,14,15);
}

// A-fragment (16x32 bf16, MxK): lane<16 holds row m, K = {0..7, 16..23};
// lane>=16 holds row m, K = {8..15, 24..31}.  p points at row + (lane<16?0:8).
__device__ __forceinline__ v16bf load_a_frag(const unsigned short* p) {
  u32x4 lo = *(const u32x4*)(p);
  u32x4 hi = *(const u32x4*)(p + 16);
  return bfcat(__builtin_bit_cast(v8bf, lo), __builtin_bit_cast(v8bf, hi));
}

// B-fragment (32x16 bf16, KxN): lane<16 holds column n = lane, K = {0..7, 16..23};
// lane>=16 holds column n = lane-16, K = {8..15, 24..31}.  p points at
// K-values of column n, i.e. base + (lane<16?0:8).
__device__ __forceinline__ v16bf load_b_frag(const unsigned short* p) {
  u32x4 lo = *(const u32x4*)(p);
  u32x4 hi = *(const u32x4*)(p + 16);
  return bfcat(__builtin_bit_cast(v8bf, lo), __builtin_bit_cast(v8bf, hi));
}

__device__ __forceinline__ v8f wmma_bf16(v16bf a, v16bf b, v8f c) {
  return __builtin_amdgcn_wmma_f32_16x16x32_bf16(
      /*neg_a=*/false, a, /*neg_b=*/false, b,
      /*c_mod=*/(short)0, c, /*reuse_a=*/false, /*reuse_b=*/false);
}

__device__ __forceinline__ unsigned short f2bf_bits(float x) {
  return __builtin_bit_cast(unsigned short, (__bf16)x);
}

// ---------------------------------------------------------------------------
// f32 -> bf16 elementwise convert (memory bound, grid-stride)
// ---------------------------------------------------------------------------
__global__ __launch_bounds__(256) void cvt_f32_to_bf16(const float* __restrict__ in,
                                                       unsigned short* __restrict__ out,
                                                       int n) {
  int i = blockIdx.x * blockDim.x + threadIdx.x;
  int stride = gridDim.x * blockDim.x;
  for (; i < n; i += stride) out[i] = f2bf_bits(in[i]);
}

// ---------------------------------------------------------------------------
// GEMM:  C[M,N] = A[M,K] (bf16 row-major) x B[N,K]^T (bf16 weights row-major)
//        + bias[N];  OUT_BF16 selects bf16 or f32 output.
// One wave computes a 32x64 tile (2x4 WMMA tiles); block = 4 waves stacked in M.
// ---------------------------------------------------------------------------
template <bool OUT_BF16>
__global__ __launch_bounds__(128) void gemm_bf16_kernel(
    const unsigned short* __restrict__ A,
    const unsigned short* __restrict__ Bw,
    const float* __restrict__ bias,
    void* __restrict__ Cout,
    int M, int N, int K) {
  const int w    = threadIdx.x >> 5;
  const int lane = threadIdx.x & 31;
  const int l15  = lane & 15;
  const bool lo_half = (lane < 16);

  const int n0 = blockIdx.x * 64;
  const int m0 = (blockIdx.y * 4 + w) * 32;

  const int akb = lo_half ? 0 : 8;    // A-frag K sub-offset
  const int bkb = lo_half ? 0 : 8;    // B-frag K sub-offset

  const unsigned short* arow0 = A + (size_t)(m0 + l15) * K + akb;
  const unsigned short* arow1 = arow0 + (size_t)16 * K;
  const unsigned short* brow[4];
#pragma unroll
  for (int nt = 0; nt < 4; ++nt)
    brow[nt] = Bw + (size_t)(n0 + nt * 16 + l15) * K + bkb;

  v8f acc[2][4];
#pragma unroll
  for (int mt = 0; mt < 2; ++mt)
#pragma unroll
    for (int nt = 0; nt < 4; ++nt) acc[mt][nt] = (v8f)0.0f;

  for (int k0 = 0; k0 < K; k0 += 32) {
    __builtin_prefetch(arow0 + k0 + 256, 0, 1);   // global_prefetch_b8
    v16bf a0 = load_a_frag(arow0 + k0);
    v16bf a1 = load_a_frag(arow1 + k0);
#pragma unroll
    for (int nt = 0; nt < 4; ++nt) {
      v16bf bf = load_b_frag(brow[nt] + k0);
      acc[0][nt] = wmma_bf16(a0, bf, acc[0][nt]);
      acc[1][nt] = wmma_bf16(a1, bf, acc[1][nt]);
    }
  }

  // Epilogue: C layout — lane holds column n = tile_n0 + (lane&15),
  // rows m = rbase + r (rbase = 0 for lanes 0..15, 8 for lanes 16..31).
  const int rbase = lo_half ? 0 : 8;
#pragma unroll
  for (int mt = 0; mt < 2; ++mt) {
#pragma unroll
    for (int nt = 0; nt < 4; ++nt) {
      const int col = n0 + nt * 16 + l15;
      const float bv = bias[col];
#pragma unroll
      for (int r = 0; r < 8; ++r) {
        const int rowm = m0 + mt * 16 + rbase + r;
        const float v = acc[mt][nt][r] + bv;
        if (OUT_BF16)
          ((unsigned short*)Cout)[(size_t)rowm * N + col] = f2bf_bits(v);
        else
          ((float*)Cout)[(size_t)rowm * N + col] = v;
      }
    }
  }
}

// ---------------------------------------------------------------------------
// TDM descriptor path.  This toolchain (amdgpu-toolchain clang-23) exposes the
// 6-arg builtin: (uint32x4 g0, int32x8 g1, int32x4 g2, int32x4 g3,
//                 int32x8 g4, i32 cpol) — extra group zero-filled.
// Stages one V chunk (32 keys x 128 dims) into LDS *transposed* ([dim][key])
// via a 3D walk: tile_dim0 = 1 elem, tile_dim1 = 32 keys (stride = row pitch),
// tile_dim2 = 128 dims (stride = 1).  LDS receives dim-major data.
// ---------------------------------------------------------------------------
#if defined(__gfx1250__) && __has_builtin(__builtin_amdgcn_tensor_load_to_lds)
#define USE_TDM 1
typedef __attribute__((ext_vector_type(4))) unsigned int tdm_u4;
typedef __attribute__((ext_vector_type(8))) int          tdm_i8;
typedef __attribute__((ext_vector_type(4))) int          tdm_i4;

__device__ __forceinline__ void tdm_load_v_transposed(const unsigned short* gptr,
                                                      unsigned lds_byte_off) {
  const unsigned long long ga = (unsigned long long)(uintptr_t)gptr;
  tdm_u4 g0;
  g0[0] = 1u;                                    // count=1, user mode
  g0[1] = lds_byte_off;                          // lds_addr
  g0[2] = (unsigned)(ga & 0xffffffffu);          // global_addr[31:0]
  g0[3] = (unsigned)((ga >> 32) & 0x01ffffffu)   // global_addr[56:32]
          | 0x80000000u;                         // type=2 (image)
  tdm_i8 g1;
  g1[0] = 0x00010000;                            // data_size = 2 bytes
  g1[1] = (int)(1u << 16);                       // tensor_dim0 = 1
  g1[2] = (int)((unsigned)S_DIM << 16);          // tensor_dim1 = 2048 keys
  g1[3] = (int)(1u << 16);                       // tile_dim0 = 1
  g1[4] = (int)(32u | (128u << 16));             // tile_dim1=32, tile_dim2=128
  g1[5] = QKV_ROW * B_DIM;                       // tensor_dim0_stride = 12288
  g1[6] = (int)(1u << 16);                       // tensor_dim1_stride = 1
  g1[7] = 0;
  tdm_i4 g2;
  g2[0] = HD_DIM;                                // tensor_dim2 = 128 dims
  g2[1] = 0; g2[2] = 0; g2[3] = 0;               // no dim3, no iterate
  tdm_i4 g3; g3[0] = 0; g3[1] = 0; g3[2] = 0; g3[3] = 0;
  tdm_i8 g4;                                     // extra group on this
  g4[0] = 0; g4[1] = 0; g4[2] = 0; g4[3] = 0;    // toolchain: zero-filled
  g4[4] = 0; g4[5] = 0; g4[6] = 0; g4[7] = 0;
  __builtin_amdgcn_tensor_load_to_lds(g0, g1, g2, g3, g4, 0);
}
#else
#define USE_TDM 0
#endif

// ---------------------------------------------------------------------------
// Flash attention: ONE WAVE per workgroup handles 16 query rows of one (b,h).
// (Single-wave blocks keep every TDM descriptor ingredient wave-uniform.)
// qkv (bf16): [S, B, NH, 384] with q = [0:128), k = [128:256), v = [256:384)
// mask (f32): [B, S, S];  ctx (bf16): [S, B, NH*HD]
// ---------------------------------------------------------------------------
__global__ __launch_bounds__(32) void flash_attn_kernel(
    const unsigned short* __restrict__ qkv,
    const float* __restrict__ mask,
    unsigned short* __restrict__ ctx) {
  __shared__ __align__(16) unsigned short vtT[2][HD_DIM][32]; // V^T, dbl-buffered
  __shared__ __align__(16) float          sc[16][32];         // score tile
  __shared__ __align__(16) unsigned short pb[16][32];         // probs (bf16)
  __shared__ __align__(16) float          corr[16];           // row factors

  const int lane = threadIdx.x & 31;
  const int l15  = lane & 15;
  const bool lo_half = (lane < 16);
  const int rbase = lo_half ? 0 : 8;

  const int bh = blockIdx.x;          // 0..B*NH-1
  const int b  = bh / NH_DIM;
  const int h  = bh % NH_DIM;
  const int q0 = blockIdx.y * 16;

  // -------- load Q fragments (16 rows x 128 dims = 4 K-chunks), keep resident
  v16bf qf[4];
  {
    const int qrowi = q0 + l15;
    const unsigned short* qrow =
        qkv + ((size_t)(qrowi * B_DIM + b) * NH_DIM + h) * 384;
    const int akb = lo_half ? 0 : 8;
#pragma unroll
    for (int c = 0; c < 4; ++c) qf[c] = load_a_frag(qrow + c * 32 + akb);
  }

  v8f acc[8];
#pragma unroll
  for (int nt = 0; nt < 8; ++nt) acc[nt] = (v8f)0.0f;

  float m_run = -1e30f;   // meaningful on lanes 0..15 (row = lane)
  float s_run = 0.0f;

  const int bkb = lo_half ? 0 : 8;

#if USE_TDM
  // prime the pipeline: DMA chunk 0 into buffer 0 while we start QK^T
  tdm_load_v_transposed(
      qkv + ((size_t)(0 * B_DIM + b) * NH_DIM + h) * 384 + 2 * HD_DIM,
      (unsigned)(uintptr_t)&vtT[0][0][0]);
#endif

  for (int t0 = 0; t0 < S_DIM; t0 += 32) {
    int bufi = (t0 >> 5) & 1;
#if USE_TDM
    if (t0 + 32 < S_DIM) {   // DMA next chunk into the other buffer
      tdm_load_v_transposed(
          qkv + ((size_t)((t0 + 32) * B_DIM + b) * NH_DIM + h) * 384 + 2 * HD_DIM,
          (unsigned)(uintptr_t)&vtT[bufi ^ 1][0][0]);
    }
#else
    bufi = 0;
    {  // manual transposed staging: lane owns key t0+lane, all 128 dims
      const unsigned short* vrow =
          qkv + ((size_t)((t0 + lane) * B_DIM + b) * NH_DIM + h) * 384 + 2 * HD_DIM;
#pragma unroll
      for (int c = 0; c < 16; ++c) {
        u32x4 v = *(const u32x4*)(vrow + c * 8);
#pragma unroll
        for (int jj = 0; jj < 4; ++jj) {
          const unsigned dw = v[jj];
          vtT[0][c * 8 + 2 * jj][lane]     = (unsigned short)(dw & 0xffffu);
          vtT[0][c * 8 + 2 * jj + 1][lane] = (unsigned short)(dw >> 16);
        }
      }
    }
#endif

    // ---- scores = Q x K^T  (two 16-key subtiles, K-dim = 128 in 4 steps)
    v8f s0 = (v8f)0.0f, s1 = (v8f)0.0f;
#pragma unroll
    for (int c = 0; c < 4; ++c) {
      const int kd = c * 32 + bkb;
      const int key0 = t0 + l15;
      const unsigned short* kp0 =
          qkv + ((size_t)(key0 * B_DIM + b) * NH_DIM + h) * 384 + HD_DIM + kd;
      const unsigned short* kp1 =
          qkv + ((size_t)((key0 + 16) * B_DIM + b) * NH_DIM + h) * 384 + HD_DIM + kd;
      v16bf kf0 = load_b_frag(kp0);
      v16bf kf1 = load_b_frag(kp1);
      s0 = wmma_bf16(qf[c], kf0, s0);
      s1 = wmma_bf16(qf[c], kf1, s1);
    }

    // ---- scale + mask, spill score tile to LDS (C layout -> row major)
    {
      const int col = l15;
#pragma unroll
      for (int r = 0; r < 8; ++r) {
        const int sq = q0 + rbase + r;
        const size_t mbase = ((size_t)b * S_DIM + sq) * S_DIM + t0;
        sc[rbase + r][col]      = s0[r] * NORM_INV + mask[mbase + col];
        sc[rbase + r][col + 16] = s1[r] * NORM_INV + mask[mbase + 16 + col];
      }
    }

    // ---- online softmax for this chunk: lane l (< 16) owns row l
    if (lane < 16) {
      float mx = m_run;
#pragma unroll
      for (int j = 0; j < 32; ++j) mx = fmaxf(mx, sc[lane][j]);
      const float cf = __expf(m_run - mx);
      float sum = s_run * cf;
#pragma unroll
      for (int j = 0; j < 32; ++j) {
        const float p = __expf(sc[lane][j] - mx);
        sum += p;
        pb[lane][j] = f2bf_bits(p);
      }
      m_run = mx;
      s_run = sum;
      corr[lane] = cf;
    }
    // (same-wave LDS ops are in-order; backend inserts s_wait_dscnt)

    // ---- rescale accumulator rows by exp(m_old - m_new)
    {
      float fr[8];
#pragma unroll
      for (int r = 0; r < 8; ++r) fr[r] = corr[rbase + r];
#pragma unroll
      for (int nt = 0; nt < 8; ++nt)
#pragma unroll
        for (int r = 0; r < 8; ++r) acc[nt][r] *= fr[r];
    }

    // ---- P fragment (A layout) from LDS probs
    v16bf pf;
    {
      const int pkb = lo_half ? 0 : 8;
      v8bf plo = __builtin_bit_cast(v8bf, *(const u32x4*)&pb[l15][pkb]);
      v8bf phi = __builtin_bit_cast(v8bf, *(const u32x4*)&pb[l15][pkb + 16]);
      pf = bfcat(plo, phi);
    }

#if USE_TDM
    // current chunk's DMA must have landed; next chunk may stay in flight
    if (t0 + 32 < S_DIM) __builtin_amdgcn_s_wait_tensorcnt(1);
    else                 __builtin_amdgcn_s_wait_tensorcnt(0);
#endif

    // ---- acc += P (16x32) x V (32x128): V^T in LDS -> pure b128 B-frags
#pragma unroll
    for (int nt = 0; nt < 8; ++nt) {
      const int nc = nt * 16 + l15;      // dim column
      const int kk = lo_half ? 0 : 8;    // key sub-range
      v16bf vf = load_b_frag(&vtT[bufi][nc][kk]);
      acc[nt] = wmma_bf16(pf, vf, acc[nt]);
    }
  }

  // ---- final normalization and bf16 store of ctx
  if (lane < 16) corr[lane] = 1.0f / s_run;
  {
    float fr[8];
#pragma unroll
    for (int r = 0; r < 8; ++r) fr[r] = corr[rbase + r];
#pragma unroll
    for (int nt = 0; nt < 8; ++nt) {
      const int n = nt * 16 + l15;
#pragma unroll
      for (int r = 0; r < 8; ++r) {
        const int sq = q0 + rbase + r;
        ctx[((size_t)(sq * B_DIM + b)) * PROJ_N + h * HD_DIM + n] =
            f2bf_bits(acc[nt][r] * fr[r]);
      }
    }
  }
}

// ---------------------------------------------------------------------------
// append b_dense (second tuple element) to d_out
// ---------------------------------------------------------------------------
__global__ __launch_bounds__(256) void copy_bias_kernel(const float* __restrict__ b,
                                                        float* __restrict__ out,
                                                        int n) {
  int i = blockIdx.x * blockDim.x + threadIdx.x;
  if (i < n) out[i] = b[i];
}

// ---------------------------------------------------------------------------
// launcher
// ---------------------------------------------------------------------------
extern "C" void kernel_launch(void* const* d_in, const int* in_sizes, int n_in,
                              void* d_out, int out_size, void* d_ws, size_t ws_size,
                              hipStream_t stream) {
  (void)in_sizes; (void)n_in; (void)out_size;

  const float* hidden  = (const float*)d_in[0];   // [S,B,H]
  const float* mask    = (const float*)d_in[1];   // [B,1,1,S,S]
  const float* w_qkv   = (const float*)d_in[2];   // [6144, 2048]
  const float* b_qkv   = (const float*)d_in[3];   // [6144]
  const float* w_dense = (const float*)d_in[4];   // [2048, 2048]
  const float* b_dense = (const float*)d_in[5];   // [2048]
  float* out = (float*)d_out;                     // output [S,B,H] ++ b_dense[H]

  // workspace layout (bf16 buffers), ~112 MB total
  char* ws = (char*)d_ws;
  unsigned short* h_bf   = (unsigned short*)(ws);
  unsigned short* wq_bf  = (unsigned short*)(ws + (size_t)M_ROWS * H_DIM * 2);
  unsigned short* wd_bf  = (unsigned short*)((char*)wq_bf + (size_t)QKV_N * H_DIM * 2);
  unsigned short* qkv_bf = (unsigned short*)((char*)wd_bf + (size_t)H_DIM * PROJ_N * 2);
  unsigned short* ctx_bf = (unsigned short*)((char*)qkv_bf + (size_t)M_ROWS * QKV_N * 2);
  (void)ws_size;  // requires ~112 MB scratch

  // 1) convert inputs to bf16
  cvt_f32_to_bf16<<<2048, 256, 0, stream>>>(hidden, h_bf, M_ROWS * H_DIM);
  cvt_f32_to_bf16<<<2048, 256, 0, stream>>>(w_qkv, wq_bf, QKV_N * H_DIM);
  cvt_f32_to_bf16<<<2048, 256, 0, stream>>>(w_dense, wd_bf, H_DIM * PROJ_N);

  // 2) QKV projection: [4096,2048] x [6144,2048]^T + b_qkv -> bf16 [4096,6144]
  {
    dim3 grid(QKV_N / 64, M_ROWS / 128);  // (96, 32)
    gemm_bf16_kernel<true><<<grid, 128, 0, stream>>>(
        h_bf, wq_bf, b_qkv, (void*)qkv_bf, M_ROWS, QKV_N, H_DIM);
  }

  // 3) flash attention -> ctx bf16 [4096, 2048]; one wave per block
  {
    dim3 grid(B_DIM * NH_DIM, S_DIM / 16);  // (32, 128)
    flash_attn_kernel<<<grid, 32, 0, stream>>>(qkv_bf, mask, ctx_bf);
  }

  // 4) output projection: ctx x w_dense^T + b_dense -> f32 d_out
  {
    dim3 grid(H_DIM / 64, M_ROWS / 128);  // (32, 32)
    gemm_bf16_kernel<false><<<grid, 128, 0, stream>>>(
        ctx_bf, wd_bf, b_dense, (void*)out, M_ROWS, H_DIM, PROJ_N);
  }

  // 5) tuple second element: append b_dense
  copy_bias_kernel<<<(H_DIM + 255) / 256, 256, 0, stream>>>(
      b_dense, out + (size_t)M_ROWS * H_DIM, H_DIM);
}
